// Q1_54116587929804
// MI455X (gfx1250) — compile-verified, matrix operand order checked by reading, not run on hardware
//
#include <hip/hip_runtime.h>
#include <hip/hip_bf16.h>
#include <cstdint>

#define NN   100000
#define DD   128
#define BB   4096
#define KK   64
#define HH   256
#define EE   128
#define DIN  129
#define DINP 160   // DIN padded to multiple of 32 for WMMA K-chunks

typedef __bf16 bf16x16 __attribute__((ext_vector_type(16)));
typedef float  floatx8 __attribute__((ext_vector_type(8)));

union AFrag { bf16x16 v; uint32_t u[8]; };
union Acc   { floatx8 v; float f[8]; };

__device__ __forceinline__ floatx8 wmma_bf16(bf16x16 a, bf16x16 b, floatx8 c) {
  return __builtin_amdgcn_wmma_f32_16x16x32_bf16(false, a, false, b, (short)0, c, false, false);
}

__device__ __forceinline__ uint32_t pack_bf16(float lo, float hi) {
  union { __bf16 b[2]; uint32_t u; } p;
  p.b[0] = (__bf16)lo; p.b[1] = (__bf16)hi;
  return p.u;
}

// A-fragment (16x32 bf16, row-major source tile in LDS) per ISA 16-bit A layout:
// lanes 0-15: M=lane, V0..V3 hold K=0..7 pairs, V4..V7 hold K=16..23 pairs
// lanes 16-31: M=lane-16, K offset +8
__device__ __forceinline__ bf16x16 load_a_lds(const __bf16* tile, int ldk,
                                              int row0, int k0, int lane) {
  AFrag a;
  int m    = row0 + (lane & 15);
  int half = lane >> 4;
  const __bf16* p = tile + m * ldk + k0 + half * 8;
  a.u[0] = *(const uint32_t*)(p + 0);
  a.u[1] = *(const uint32_t*)(p + 2);
  a.u[2] = *(const uint32_t*)(p + 4);
  a.u[3] = *(const uint32_t*)(p + 6);
  a.u[4] = *(const uint32_t*)(p + 16);
  a.u[5] = *(const uint32_t*)(p + 18);
  a.u[6] = *(const uint32_t*)(p + 20);
  a.u[7] = *(const uint32_t*)(p + 22);
  return a.v;
}

// B-fragment from pre-swizzled weight blocks: contiguous 32B per lane.
__device__ __forceinline__ bf16x16 load_b_blk(const uint32_t* blocks, int NT,
                                              int kc, int nt, int lane) {
  AFrag b;
  const uint4* q = (const uint4*)(blocks + ((kc * NT + nt) << 8) + (lane << 3));
  uint4 x0 = q[0], x1 = q[1];
  b.u[0] = x0.x; b.u[1] = x0.y; b.u[2] = x0.z; b.u[3] = x0.w;
  b.u[4] = x1.x; b.u[5] = x1.y; b.u[6] = x1.z; b.u[7] = x1.w;
  return b.v;
}

// ---------------------------------------------------------------------------
// Weight swizzle: W row-major [Nsrc][Ksrc] fp32 -> bf16 fragment blocks.
// dst[(kc*NT+nt)*256 + lane*8 + v] packs (WT[k][n], WT[k+1][n]) with the exact
// lane/VGPR mapping the WMMA B operand wants (zero-padded past Ksrc/Nsrc).
// ---------------------------------------------------------------------------
__global__ void swizzle_weight(const float* __restrict__ W, uint32_t* __restrict__ dst,
                               int Ksrc, int Nsrc, int KC, int NT) {
  int idx = blockIdx.x * blockDim.x + threadIdx.x;
  int total = KC * NT * 256;
  if (idx >= total) return;
  int v    = idx & 7;
  int lane = (idx >> 3) & 31;
  int blk  = idx >> 8;
  int nt   = blk % NT;
  int kc   = blk / NT;
  int k = kc * 32 + ((v < 4) ? 2 * v : 16 + 2 * (v - 4)) + ((lane >= 16) ? 8 : 0);
  int n = nt * 16 + (lane & 15);
  float lo = (k     < Ksrc && n < Nsrc) ? W[n * Ksrc + k]     : 0.0f;
  float hi = (k + 1 < Ksrc && n < Nsrc) ? W[n * Ksrc + k + 1] : 0.0f;
  dst[idx] = pack_bf16(lo, hi);
}

__global__ void prep_misc(const float* __restrict__ bih, const float* __restrict__ bhh,
                          float* __restrict__ biasg,
                          const float* __restrict__ h0, __bf16* __restrict__ h0b) {
  int idx = blockIdx.x * blockDim.x + threadIdx.x;
  if (idx < 4 * HH) biasg[idx] = bih[idx] + bhh[idx];
  if (idx < BB * HH) h0b[idx] = (__bf16)h0[idx];
}

// ---------------------------------------------------------------------------
// K1: x = relu([feat(actions0), step] @ W1a^T + b1a) @ W1b^T + b1b  (bf16 out)
// 64 rows / block, 8 waves. GEMM1 -> LDS hidden -> GEMM2.
// ---------------------------------------------------------------------------
__global__ __launch_bounds__(256) void action_mlp1(
    const float* __restrict__ node, const float* __restrict__ steps,
    const int* __restrict__ actions0,
    const float* __restrict__ b1a, const float* __restrict__ b1b,
    const uint32_t* __restrict__ w1aT, const uint32_t* __restrict__ w1bT,
    __bf16* __restrict__ xout) {
  __shared__ __bf16 aT[64 * DINP];   // 20 KB
  __shared__ __bf16 hid[64 * 256];   // 32 KB
  __shared__ int    aid[64];
  int tid = threadIdx.x;
  int gb0 = blockIdx.x * 64;
  if (tid < 64) aid[tid] = actions0[gb0 + tid];
  __syncthreads();
  for (int idx = tid; idx < 64 * DINP; idx += 256) {
    int r = idx / DINP, c = idx - r * DINP;
    float v;
    if (c < DD)       v = node[(size_t)aid[r] * DD + c];
    else if (c == DD) v = steps[gb0 + r] * 0.01f;
    else              v = 0.0f;
    aT[idx] = (__bf16)v;
  }
  __syncthreads();

  int lane = tid & 31, w = tid >> 5;
  int rt = w >> 1, nh = w & 1;
  int half = lane >> 4, mcol = lane & 15;

  Acc acc[8];
#pragma unroll
  for (int i = 0; i < 8; ++i)
#pragma unroll
    for (int r = 0; r < 8; ++r) acc[i].f[r] = 0.0f;
  for (int kc = 0; kc < 5; ++kc) {
    bf16x16 a = load_a_lds(aT, DINP, rt * 16, kc * 32, lane);
#pragma unroll
    for (int i = 0; i < 8; ++i) {
      bf16x16 b = load_b_blk(w1aT, 16, kc, nh * 8 + i, lane);
      acc[i].v = wmma_bf16(a, b, acc[i].v);
    }
  }
#pragma unroll
  for (int i = 0; i < 8; ++i) {
    int col = (nh * 8 + i) * 16 + mcol;
    float bb = b1a[col];
#pragma unroll
    for (int r = 0; r < 8; ++r) {
      int row = rt * 16 + half * 8 + r;
      float v = acc[i].f[r] + bb;
      hid[row * 256 + col] = (__bf16)(v > 0.0f ? v : 0.0f);
    }
  }
  __syncthreads();

  Acc o[4];
#pragma unroll
  for (int i = 0; i < 4; ++i)
#pragma unroll
    for (int r = 0; r < 8; ++r) o[i].f[r] = 0.0f;
  for (int kc = 0; kc < 8; ++kc) {
    bf16x16 a = load_a_lds(hid, 256, rt * 16, kc * 32, lane);
#pragma unroll
    for (int i = 0; i < 4; ++i) {
      bf16x16 b = load_b_blk(w1bT, 8, kc, nh * 4 + i, lane);
      o[i].v = wmma_bf16(a, b, o[i].v);
    }
  }
#pragma unroll
  for (int i = 0; i < 4; ++i) {
    int col = (nh * 4 + i) * 16 + mcol;
    float bb = b1b[col];
#pragma unroll
    for (int r = 0; r < 8; ++r) {
      int row = gb0 + rt * 16 + half * 8 + r;
      xout[(size_t)row * 128 + col] = (__bf16)(o[i].f[r] + bb);
    }
  }
}

// ---------------------------------------------------------------------------
// K2a: gates = x @ Wih^T + h0 @ Whh^T + (bih+bhh)   [B,1024] fp32
// ---------------------------------------------------------------------------
__global__ __launch_bounds__(256) void gates_gemm(
    const __bf16* __restrict__ xin, const __bf16* __restrict__ h0b,
    const uint32_t* __restrict__ wihT, const uint32_t* __restrict__ whhT,
    const float* __restrict__ biasg, float* __restrict__ gates) {
  __shared__ __bf16 xT[64 * 128];    // 16 KB
  __shared__ __bf16 hT[64 * 256];    // 32 KB
  int tid = threadIdx.x;
  int gb0 = blockIdx.x * 64;
  {
    const uint32_t* sx = (const uint32_t*)(xin + (size_t)gb0 * 128);
    const uint32_t* sh = (const uint32_t*)(h0b + (size_t)gb0 * 256);
    uint32_t* dx = (uint32_t*)xT;
    uint32_t* dh = (uint32_t*)hT;
    for (int i = tid; i < 64 * 128 / 2; i += 256) dx[i] = sx[i];
    for (int i = tid; i < 64 * 256 / 2; i += 256) dh[i] = sh[i];
  }
  __syncthreads();

  int lane = tid & 31, w = tid >> 5;
  int rt = w >> 1, nh = w & 1;
  int half = lane >> 4, mcol = lane & 15;

  for (int ng = 0; ng < 4; ++ng) {
    Acc acc[8];
#pragma unroll
    for (int i = 0; i < 8; ++i)
#pragma unroll
      for (int r = 0; r < 8; ++r) acc[i].f[r] = 0.0f;
    for (int kc = 0; kc < 4; ++kc) {
      bf16x16 a = load_a_lds(xT, 128, rt * 16, kc * 32, lane);
#pragma unroll
      for (int i = 0; i < 8; ++i) {
        bf16x16 b = load_b_blk(wihT, 64, kc, ng * 16 + nh * 8 + i, lane);
        acc[i].v = wmma_bf16(a, b, acc[i].v);
      }
    }
    for (int kc = 0; kc < 8; ++kc) {
      bf16x16 a = load_a_lds(hT, 256, rt * 16, kc * 32, lane);
#pragma unroll
      for (int i = 0; i < 8; ++i) {
        bf16x16 b = load_b_blk(whhT, 64, kc, ng * 16 + nh * 8 + i, lane);
        acc[i].v = wmma_bf16(a, b, acc[i].v);
      }
    }
#pragma unroll
    for (int i = 0; i < 8; ++i) {
      int col = (ng * 16 + nh * 8 + i) * 16 + mcol;
      float bb = biasg[col];
#pragma unroll
      for (int r = 0; r < 8; ++r) {
        int row = gb0 + rt * 16 + half * 8 + r;
        gates[(size_t)row * 1024 + col] = acc[i].f[r] + bb;
      }
    }
  }
}

// ---------------------------------------------------------------------------
// K2b: LSTM nonlinearity -> h (out), c (out), h bf16 (ws)
// ---------------------------------------------------------------------------
__global__ void lstm_elem(const float* __restrict__ gates, const float* __restrict__ c0,
                          float* __restrict__ hout, float* __restrict__ cout,
                          __bf16* __restrict__ hb) {
  int idx = blockIdx.x * blockDim.x + threadIdx.x;
  if (idx >= BB * HH) return;
  int b = idx / HH, j = idx - b * HH;
  const float* g = gates + (size_t)b * 1024;
  float gi = g[j], gf = g[256 + j], gg = g[512 + j], go = g[768 + j];
  float si = 1.0f / (1.0f + __expf(-gi));
  float sf = 1.0f / (1.0f + __expf(-gf));
  float so = 1.0f / (1.0f + __expf(-go));
  float c = sf * c0[idx] + si * tanhf(gg);
  float h = so * tanhf(c);
  hout[idx] = h;
  cout[idx] = c;
  hb[idx] = (__bf16)h;
}

// ---------------------------------------------------------------------------
// K3: emb = h @ W3^T + b3   [B,128] fp32
// ---------------------------------------------------------------------------
__global__ __launch_bounds__(256) void emb_gemm(
    const __bf16* __restrict__ hb, const uint32_t* __restrict__ w3T,
    const float* __restrict__ b3, float* __restrict__ emb) {
  __shared__ __bf16 hT[64 * 256];
  int tid = threadIdx.x;
  int gb0 = blockIdx.x * 64;
  {
    const uint32_t* sh = (const uint32_t*)(hb + (size_t)gb0 * 256);
    uint32_t* dh = (uint32_t*)hT;
    for (int i = tid; i < 64 * 256 / 2; i += 256) dh[i] = sh[i];
  }
  __syncthreads();
  int lane = tid & 31, w = tid >> 5;
  int rt = w >> 1, nh = w & 1;
  int half = lane >> 4, mcol = lane & 15;
  Acc o[4];
#pragma unroll
  for (int i = 0; i < 4; ++i)
#pragma unroll
    for (int r = 0; r < 8; ++r) o[i].f[r] = 0.0f;
  for (int kc = 0; kc < 8; ++kc) {
    bf16x16 a = load_a_lds(hT, 256, rt * 16, kc * 32, lane);
#pragma unroll
    for (int i = 0; i < 4; ++i) {
      bf16x16 b = load_b_blk(w3T, 8, kc, nh * 4 + i, lane);
      o[i].v = wmma_bf16(a, b, o[i].v);
    }
  }
#pragma unroll
  for (int i = 0; i < 4; ++i) {
    int col = (nh * 4 + i) * 16 + mcol;
    float bb = b3[col];
#pragma unroll
    for (int r = 0; r < 8; ++r) {
      int row = gb0 + rt * 16 + half * 8 + r;
      emb[(size_t)row * 128 + col] = o[i].f[r] + bb;
    }
  }
}

// ---------------------------------------------------------------------------
// K4: fused neighbor MLP2 + masked scaled dot (never materializes g[B,K,E]).
// One block per batch row b; its K=64 neighbors are the 64 GEMM rows.
// ---------------------------------------------------------------------------
__global__ __launch_bounds__(256) void neighbor_fused(
    const float* __restrict__ node, const float* __restrict__ steps,
    const int* __restrict__ neighbors, const int* __restrict__ deg,
    const float* __restrict__ b2a, const float* __restrict__ b2b,
    const uint32_t* __restrict__ w2aT, const uint32_t* __restrict__ w2bT,
    const float* __restrict__ emb, float* __restrict__ vals) {
  __shared__ __bf16 aT[64 * DINP];   // 20 KB
  __shared__ __bf16 hid[64 * 256];   // 32 KB
  __shared__ float  embs[128];
  __shared__ float  vacc[64];
  __shared__ int    nid[64];
  int tid = threadIdx.x;
  int b = blockIdx.x;
  float stepv = steps[b] * 0.01f;
  if (tid < 128) embs[tid] = emb[(size_t)b * 128 + tid];
  if (tid < 64) { vacc[tid] = 0.0f; nid[tid] = neighbors[b * KK + tid]; }
  __syncthreads();
  for (int idx = tid; idx < 64 * DINP; idx += 256) {
    int r = idx / DINP, c = idx - r * DINP;
    float v;
    if (c < DD)       v = node[(size_t)nid[r] * DD + c];
    else if (c == DD) v = stepv;
    else              v = 0.0f;
    aT[idx] = (__bf16)v;
  }
  __syncthreads();

  int lane = tid & 31, w = tid >> 5;
  int rt = w >> 1, nh = w & 1;
  int half = lane >> 4, mcol = lane & 15;

  // layer 1: [64 x 160] @ [160 x 256] + b2a, relu
  Acc acc[8];
#pragma unroll
  for (int i = 0; i < 8; ++i)
#pragma unroll
    for (int r = 0; r < 8; ++r) acc[i].f[r] = 0.0f;
  for (int kc = 0; kc < 5; ++kc) {
    bf16x16 a = load_a_lds(aT, DINP, rt * 16, kc * 32, lane);
#pragma unroll
    for (int i = 0; i < 8; ++i) {
      bf16x16 bB = load_b_blk(w2aT, 16, kc, nh * 8 + i, lane);
      acc[i].v = wmma_bf16(a, bB, acc[i].v);
    }
  }
#pragma unroll
  for (int i = 0; i < 8; ++i) {
    int col = (nh * 8 + i) * 16 + mcol;
    float bb = b2a[col];
#pragma unroll
    for (int r = 0; r < 8; ++r) {
      int row = rt * 16 + half * 8 + r;
      float v = acc[i].f[r] + bb;
      hid[row * 256 + col] = (__bf16)(v > 0.0f ? v : 0.0f);
    }
  }
  __syncthreads();

  // layer 2: [64 x 256] @ [256 x 128] + b2b, fused dot with emb[b]
  Acc o[4];
#pragma unroll
  for (int i = 0; i < 4; ++i)
#pragma unroll
    for (int r = 0; r < 8; ++r) o[i].f[r] = 0.0f;
  for (int kc = 0; kc < 8; ++kc) {
    bf16x16 a = load_a_lds(hid, 256, rt * 16, kc * 32, lane);
#pragma unroll
    for (int i = 0; i < 4; ++i) {
      bf16x16 bB = load_b_blk(w2bT, 8, kc, nh * 4 + i, lane);
      o[i].v = wmma_bf16(a, bB, o[i].v);
    }
  }
  float p[8];
#pragma unroll
  for (int r = 0; r < 8; ++r) p[r] = 0.0f;
#pragma unroll
  for (int i = 0; i < 4; ++i) {
    int col = (nh * 4 + i) * 16 + mcol;
    float bb = b2b[col];
    float ev = embs[col];
#pragma unroll
    for (int r = 0; r < 8; ++r) p[r] += (o[i].f[r] + bb) * ev;
  }
#pragma unroll
  for (int r = 0; r < 8; ++r)
    atomicAdd(&vacc[rt * 16 + half * 8 + r], p[r]);
  __syncthreads();

  if (tid < 64) {
    int dg = deg[b];
    if (dg < 1) dg = 1;
    float v = (tid < dg) ? vacc[tid] * 0.08838834764831845f : 0.0f; // 1/sqrt(128)
    vals[b * KK + tid] = v;
  }
}

// ---------------------------------------------------------------------------
extern "C" void kernel_launch(void* const* d_in, const int* in_sizes, int n_in,
                              void* d_out, int out_size, void* d_ws, size_t ws_size,
                              hipStream_t stream) {
  const float* node  = (const float*)d_in[0];
  const float* steps = (const float*)d_in[1];
  const float* h0    = (const float*)d_in[2];
  const float* c0    = (const float*)d_in[3];
  const float* W1a   = (const float*)d_in[4];
  const float* b1a   = (const float*)d_in[5];
  const float* W1b   = (const float*)d_in[6];
  const float* b1b   = (const float*)d_in[7];
  const float* W2a   = (const float*)d_in[8];
  const float* b2a   = (const float*)d_in[9];
  const float* W2b   = (const float*)d_in[10];
  const float* b2b   = (const float*)d_in[11];
  const float* W3    = (const float*)d_in[12];
  const float* b3    = (const float*)d_in[13];
  const float* Wih   = (const float*)d_in[14];
  const float* Whh   = (const float*)d_in[15];
  const float* bih   = (const float*)d_in[16];
  const float* bhh   = (const float*)d_in[17];
  const int* actions0  = (const int*)d_in[18];
  const int* neighbors = (const int*)d_in[19];
  const int* deg       = (const int*)d_in[20];

  float* out  = (float*)d_out;
  float* vals = out;                       // [B*K]
  float* hout = out + BB * KK;             // [B*H]
  float* cout = out + BB * KK + BB * HH;   // [B*H]

  char* ws = (char*)d_ws;
  size_t off = 0;
  auto alloc = [&](size_t bytes) -> void* {
    void* p = ws + off;
    off += (bytes + 255) & ~(size_t)255;
    return p;
  };
  uint32_t* w1aT  = (uint32_t*)alloc(5 * 16 * 256 * 4);
  uint32_t* w1bT  = (uint32_t*)alloc(8 * 8 * 256 * 4);
  uint32_t* wihT  = (uint32_t*)alloc(4 * 64 * 256 * 4);
  uint32_t* whhT  = (uint32_t*)alloc(8 * 64 * 256 * 4);
  uint32_t* w3T   = (uint32_t*)alloc(8 * 8 * 256 * 4);
  uint32_t* w2aT  = (uint32_t*)alloc(5 * 16 * 256 * 4);
  uint32_t* w2bT  = (uint32_t*)alloc(8 * 8 * 256 * 4);
  float*    biasg = (float*)alloc(4 * HH * 4);
  __bf16*   h0b   = (__bf16*)alloc((size_t)BB * HH * 2);
  __bf16*   xb    = (__bf16*)alloc((size_t)BB * 128 * 2);
  float*    gates = (float*)alloc((size_t)BB * 1024 * 4);
  __bf16*   hb    = (__bf16*)alloc((size_t)BB * HH * 2);
  float*    embw  = (float*)alloc((size_t)BB * 128 * 4);

  auto gridFor = [](int total) { return dim3((total + 255) / 256); };

  // weight swizzles (bf16 fragment-native layout)
  swizzle_weight<<<gridFor(5 * 16 * 256), 256, 0, stream>>>(W1a, w1aT, DIN, 256, 5, 16);
  swizzle_weight<<<gridFor(8 * 8 * 256), 256, 0, stream>>>(W1b, w1bT, 256, 128, 8, 8);
  swizzle_weight<<<gridFor(4 * 64 * 256), 256, 0, stream>>>(Wih, wihT, 128, 1024, 4, 64);
  swizzle_weight<<<gridFor(8 * 64 * 256), 256, 0, stream>>>(Whh, whhT, 256, 1024, 8, 64);
  swizzle_weight<<<gridFor(8 * 8 * 256), 256, 0, stream>>>(W3, w3T, 256, 128, 8, 8);
  swizzle_weight<<<gridFor(5 * 16 * 256), 256, 0, stream>>>(W2a, w2aT, DIN, 256, 5, 16);
  swizzle_weight<<<gridFor(8 * 8 * 256), 256, 0, stream>>>(W2b, w2bT, 256, 128, 8, 8);
  prep_misc<<<gridFor(BB * HH), 256, 0, stream>>>(bih, bhh, biasg, h0, h0b);

  // action path
  action_mlp1<<<dim3(BB / 64), 256, 0, stream>>>(node, steps, actions0, b1a, b1b,
                                                 w1aT, w1bT, xb);
  gates_gemm<<<dim3(BB / 64), 256, 0, stream>>>(xb, h0b, wihT, whhT, biasg, gates);
  lstm_elem<<<gridFor(BB * HH), 256, 0, stream>>>(gates, c0, hout, cout, hb);
  emb_gemm<<<dim3(BB / 64), 256, 0, stream>>>(hb, w3T, b3, embw);

  // fused neighbor MLP + masked scaled dot
  neighbor_fused<<<dim3(BB), 256, 0, stream>>>(node, steps, neighbors, deg,
                                               b2a, b2b, w2aT, w2bT, embw, vals);
}
